// SpeculateGRUCell_54065048322168
// MI455X (gfx1250) — compile-verified, matrix operand order checked by reading, not run on hardware
//
#include <hip/hip_runtime.h>
#include <math.h>

#define BN 128
#define NN 1024
#define DD 16
#define UU 128
#define FF 144            // D + U
#define KK 160            // FF zero-padded to a multiple of 32
#define MM (BN * NN)      // 131072 rows when (b,n) flattened
#define KPAD 56           // LDS pitch in bf16: 112B = 7*16 (b128-aligned),
                          // m*28 mod 64 distinct => conflict-free frag reads

#if defined(__gfx1250__) && __has_builtin(__builtin_amdgcn_global_load_async_to_lds_b128)
#define HAVE_ASYNC 1
#endif

typedef __attribute__((ext_vector_type(16))) __bf16       v16bf;
typedef __attribute__((ext_vector_type(8)))  float        v8f;
typedef __attribute__((ext_vector_type(8)))  unsigned int v8u;
typedef __attribute__((ext_vector_type(4)))  int          i32x4;

union BFrag { v8u u; v16bf h; };

__device__ __forceinline__ unsigned short f2bf(float f) {
  unsigned int u = __builtin_bit_cast(unsigned int, f);
  u += 0x7FFFu + ((u >> 16) & 1u);          // round-to-nearest-even
  return (unsigned short)(u >> 16);
}
__device__ __forceinline__ float bf2f(unsigned short h) {
  unsigned int u = ((unsigned int)h) << 16;
  return __builtin_bit_cast(float, u);
}
__device__ __forceinline__ float sigmoidf_(float x) {
  return 1.0f / (1.0f + expf(-x));
}

// 16B global -> LDS copy. Async path: GLOBAL_LOAD_ASYNC_TO_LDS_B128, no data
// VGPRs, tracked by ASYNCcnt (overlaps WMMA). Fallback: register round-trip.
__device__ __forceinline__ void cp16_g2l(void* lds, const void* g) {
#ifdef HAVE_ASYNC
  __builtin_amdgcn_global_load_async_to_lds_b128(
      (__attribute__((address_space(1))) i32x4*)g,
      (__attribute__((address_space(3))) i32x4*)lds, 0, 0);
#else
  *(uint4*)lds = *(const uint4*)g;
#endif
}
__device__ __forceinline__ void async_wait0() {
#ifdef HAVE_ASYNC
#if __has_builtin(__builtin_amdgcn_s_wait_asynccnt)
  __builtin_amdgcn_s_wait_asynccnt(0);
#else
  asm volatile("s_wait_asynccnt 0x0" ::: "memory");
#endif
#endif
}

// 16x32 bf16 fragment from an LDS tile row (pitch KPAD). Per CDNA5 ISA
// 7.12.2: lane-half h takes K 8h..8h+7 and 16+8h..16+8h+7 (8+8).
__device__ __forceinline__ v16bf load_frag(const unsigned short* p, int half) {
  BFrag f;
  uint4 lo = *(const uint4*)(p + half * 8);
  uint4 hi = *(const uint4*)(p + 16 + half * 8);
  f.u[0] = lo.x; f.u[1] = lo.y; f.u[2] = lo.z; f.u[3] = lo.w;
  f.u[4] = hi.x; f.u[5] = hi.y; f.u[6] = hi.z; f.u[7] = hi.w;
  return f.h;
}

__device__ __forceinline__ v8f wmma_bf16(v16bf a, v16bf b, v8f c) {
  return __builtin_amdgcn_wmma_f32_16x16x32_bf16(false, a, false, b,
                                                 (short)0, c, false, false);
}

// ---------------------------------------------------------------- prep ----
__global__ __launch_bounds__(256) void k_prep_support(
    const float* __restrict__ s, unsigned short* __restrict__ S16) {
  int idx = blockIdx.x * blockDim.x + threadIdx.x;
  if (idx < NN * NN) S16[idx] = f2bf(s[idx]);
}

// xsT[b][c][n] : c<16 -> y_basis[b][n][c], else hx[b][n*128 + c-16]
__global__ __launch_bounds__(256) void k_prep_xs(
    const float* __restrict__ ybasis, const float* __restrict__ hx,
    unsigned short* __restrict__ xs1T, unsigned short* __restrict__ xs2T) {
  int idx = blockIdx.x * blockDim.x + threadIdx.x;
  if (idx >= BN * FF * NN) return;
  int n = idx & (NN - 1);
  int c = (idx >> 10) % FF;
  int b = idx / (FF * NN);
  float v;
  if (c < DD) v = ybasis[((size_t)(b * NN + n)) * DD + c];
  else        v = hx[(size_t)b * NN * UU + (size_t)n * UU + (c - DD)];
  unsigned short h = f2bf(v);
  xs1T[idx] = h;
  if (c < DD) xs2T[idx] = h;   // gconv2 shares the y_basis columns
}

// weights -> transposed, K-padded bf16: wT[c][k], zero for k>=144
__global__ __launch_bounds__(256) void k_prep_weights(
    const float* __restrict__ kr, const float* __restrict__ kc,
    const float* __restrict__ wo, unsigned short* __restrict__ krT,
    unsigned short* __restrict__ kcT, unsigned short* __restrict__ woT) {
  int idx = blockIdx.x * blockDim.x + threadIdx.x;
  if (idx < 256 * KK) {
    int c = idx / KK, k = idx % KK;
    krT[idx] = f2bf((k < FF) ? kr[(size_t)k * 256 + c] : 0.0f);
  } else if (idx < (256 + 128) * KK) {
    int i2 = idx - 256 * KK, c = i2 / KK, k = i2 % KK;
    kcT[i2] = f2bf((k < FF) ? kc[(size_t)k * 128 + c] : 0.0f);
  } else if (idx < (256 + 128 + 16) * KK) {
    int i2 = idx - (256 + 128) * KK, c = i2 / KK, k = i2 % KK;
    woT[i2] = f2bf((k < FF) ? wo[(size_t)k * DD + c] : 0.0f);
  }
}

__global__ __launch_bounds__(256) void k_new_delta(
    const float* __restrict__ yres, const float* __restrict__ delta,
    const float* __restrict__ w_basis, const float* __restrict__ b_basis,
    const float* __restrict__ lam, float* __restrict__ nd) {
  int idx = blockIdx.x * blockDim.x + threadIdx.x;
  if (idx >= MM * DD) return;
  int row = idx >> 4, d = idx & 15;
  float yr = yres[idx];
  float dl = delta[idx];
  float pt = sigmoidf_(yr + 0.5f * sqrtf(fabsf(dl + 1e-9f)));
  float s = b_basis[d];
#pragma unroll
  for (int j = 0; j < DD; ++j)
    s += yres[(size_t)row * DD + j] * w_basis[j * DD + d];
  s = fmaxf(s, 0.0f);
  nd[idx] = lam[0] * (2.0f * pt - 1.0f) * s;
}

// ------------------------------------------------------- support GEMM ----
// T16[b] = bf16( support(1024x1024) @ xs[b](1024x144) ), stored with pitch
// KK=160 and zeroed pad columns so downstream A-tiles are raw 16B copies.
// 128-row M-tile per block (halves B-side L2 traffic vs 64). 4 waves; each
// wave owns 32 rows = 2 row-tiles x 9 col-tiles = 18 f32 accumulators.
// Double-buffered LDS filled by async-to-LDS DMA; ASYNCcnt wait lands after
// the 18 WMMAs. Async issue is branch-free except one tid<64 tail.
__global__ __launch_bounds__(128, 1) void k_gconv(
    const unsigned short* __restrict__ S16,
    const unsigned short* __restrict__ xsT,
    unsigned short* __restrict__ T16) {
  __shared__ __align__(16) unsigned short sA[2][128 * KPAD];
  __shared__ __align__(16) unsigned short sB[2][FF * KPAD];
  const int b    = blockIdx.y;
  const int m0   = blockIdx.x * 128;
  const int tid  = threadIdx.x;
  const int lane = tid & 31;
  const int wave = tid >> 5;
  const int lm   = lane & 15;
  const int half = lane >> 4;
  const unsigned short* Sb  = S16 + (size_t)m0 * NN;
  const unsigned short* xsb = xsT + (size_t)b * FF * NN;

  // 16B chunks: A tile 128x32 = 512 (4/thread, uniform),
  //             B tile 144x32 = 576 (4/thread uniform + tid<64 tail)
  auto issue_async = [&](int k0, int buf) {
#pragma unroll
    for (int j = 0; j < 4; ++j) {                 // A rows 0..127
      int ch = tid + j * 128, r = ch >> 2, p = ch & 3;
      cp16_g2l(&sA[buf][r * KPAD + p * 8], &Sb[(size_t)r * NN + k0 + p * 8]);
    }
#pragma unroll
    for (int j = 0; j < 4; ++j) {                 // B rows 0..127
      int ch = tid + j * 128, r = ch >> 2, p = ch & 3;
      cp16_g2l(&sB[buf][r * KPAD + p * 8], &xsb[(size_t)r * NN + k0 + p * 8]);
    }
    if (tid < 64) {                               // B rows 128..143
      int r = 128 + (tid >> 2), p = tid & 3;
      cp16_g2l(&sB[buf][r * KPAD + p * 8], &xsb[(size_t)r * NN + k0 + p * 8]);
    }
  };

  v8f acc[18];
  v8f z = {};
#pragma unroll
  for (int t = 0; t < 18; ++t) acc[t] = z;

  issue_async(0, 0);
  async_wait0();
  __syncthreads();

  int buf = 0;
  for (int k0 = 0; k0 < NN; k0 += 32) {
    if (k0 + 32 < NN) issue_async(k0 + 32, buf ^ 1);   // DMA next chunk
    v16bf a0 = load_frag(&sA[buf][(wave * 32 + lm) * KPAD], half);
    v16bf a1 = load_frag(&sA[buf][(wave * 32 + 16 + lm) * KPAD], half);
#pragma unroll
    for (int t = 0; t < 9; ++t) {
      v16bf bf = load_frag(&sB[buf][(t * 16 + lm) * KPAD], half);
      acc[t]     = wmma_bf16(a0, bf, acc[t]);
      acc[t + 9] = wmma_bf16(a1, bf, acc[t + 9]);
    }
    if (k0 + 32 < NN) async_wait0();
    __syncthreads();
    buf ^= 1;
  }

  unsigned short* Tb = T16 + ((size_t)b * NN + m0 + wave * 32) * KK;
#pragma unroll
  for (int g = 0; g < 2; ++g) {
#pragma unroll
    for (int t = 0; t < 9; ++t)
#pragma unroll
      for (int r = 0; r < 8; ++r)
        Tb[(size_t)(g * 16 + r + 8 * half) * KK + t * 16 + lm] =
            f2bf(acc[g * 9 + t][r]);
#pragma unroll
    for (int r = 0; r < 8; ++r)                 // zero the K-pad columns
      Tb[(size_t)(g * 16 + r + 8 * half) * KK + FF + lm] = 0;
  }
}

// --------------------------------------------- gate r/u projection GEMM ----
// value = sigmoid(T1 @ kernel_r)  (K padded to 160, Ncols=256)
// Epilogue: r*hx -> xs2T columns 16..143 (bf16), u -> u16 (bf16).
__global__ __launch_bounds__(256) void k_gate_r(
    const unsigned short* __restrict__ T16,
    const unsigned short* __restrict__ krT, const float* __restrict__ hx,
    unsigned short* __restrict__ xs2T, unsigned short* __restrict__ u16) {
  __shared__ __align__(16) unsigned short sA[64 * KPAD];
  __shared__ __align__(16) unsigned short sB[256 * KPAD];
  const int m0 = blockIdx.x * 64;
  const int tid = threadIdx.x;
  const int lane = tid & 31, wave = tid >> 5;
  const int rg = wave & 3, cg = wave >> 2;       // 4 row groups x 2 col groups
  const int lm = lane & 15, half = lane >> 4;

  v8f acc[8];
  v8f z = {};
#pragma unroll
  for (int t = 0; t < 8; ++t) acc[t] = z;

  for (int k0 = 0; k0 < KK; k0 += 32) {
    {   // A-tile: 256 chunks, 1/thread
      int r = tid >> 2, p = tid & 3;
      cp16_g2l(&sA[r * KPAD + p * 8], &T16[(size_t)(m0 + r) * KK + k0 + p * 8]);
    }
#pragma unroll
    for (int j = 0; j < 4; ++j) {   // B-tile: 1024 chunks, 4/thread
      int ch = tid + j * 256, c = ch >> 2, p = ch & 3;
      cp16_g2l(&sB[c * KPAD + p * 8], &krT[(size_t)c * KK + k0 + p * 8]);
    }
    async_wait0();
    __syncthreads();
    v16bf a = load_frag(&sA[(rg * 16 + lm) * KPAD], half);
#pragma unroll
    for (int t = 0; t < 8; ++t) {
      v16bf bf = load_frag(&sB[((cg * 8 + t) * 16 + lm) * KPAD], half);
      acc[t] = wmma_bf16(a, bf, acc[t]);
    }
    __syncthreads();
  }
#pragma unroll
  for (int t = 0; t < 8; ++t) {
    int col = (cg * 8 + t) * 16 + lm;
#pragma unroll
    for (int r = 0; r < 8; ++r) {
      int row = m0 + rg * 16 + r + 8 * half;
      float v = sigmoidf_(acc[t][r]);
      if (col < UU) {                 // r-gate: write r*hx as gconv2 operand
        int bb = row >> 10, nr = row & (NN - 1);
        float h = hx[(size_t)row * UU + col];
        xs2T[((size_t)bb * FF + DD + col) * NN + nr] = f2bf(v * h);
      } else {                        // u-gate
        u16[(size_t)row * UU + (col - UU)] = f2bf(v);
      }
    }
  }
}

// ------------------------------------------------- candidate + GRU blend ---
// c = tanh(T2 @ kernel_c); new_state = hx*u + c*(1-u)
__global__ __launch_bounds__(256) void k_gate_c(
    const unsigned short* __restrict__ T16,
    const unsigned short* __restrict__ kcT, const float* __restrict__ hx,
    const unsigned short* __restrict__ u16, float* __restrict__ ns) {
  __shared__ __align__(16) unsigned short sA[64 * KPAD];
  __shared__ __align__(16) unsigned short sB[128 * KPAD];
  const int m0 = blockIdx.x * 64;
  const int tid = threadIdx.x;
  const int lane = tid & 31, wave = tid >> 5;
  const int rg = wave & 3, cg = wave >> 2;       // 2 col groups x 4 tiles
  const int lm = lane & 15, half = lane >> 4;

  v8f acc[4];
  v8f z = {};
#pragma unroll
  for (int t = 0; t < 4; ++t) acc[t] = z;

  for (int k0 = 0; k0 < KK; k0 += 32) {
    {
      int r = tid >> 2, p = tid & 3;
      cp16_g2l(&sA[r * KPAD + p * 8], &T16[(size_t)(m0 + r) * KK + k0 + p * 8]);
    }
#pragma unroll
    for (int j = 0; j < 2; ++j) {   // B-tile: 512 chunks, 2/thread
      int ch = tid + j * 256, c = ch >> 2, p = ch & 3;
      cp16_g2l(&sB[c * KPAD + p * 8], &kcT[(size_t)c * KK + k0 + p * 8]);
    }
    async_wait0();
    __syncthreads();
    v16bf a = load_frag(&sA[(rg * 16 + lm) * KPAD], half);
#pragma unroll
    for (int t = 0; t < 4; ++t) {
      v16bf bf = load_frag(&sB[((cg * 4 + t) * 16 + lm) * KPAD], half);
      acc[t] = wmma_bf16(a, bf, acc[t]);
    }
    __syncthreads();
  }
#pragma unroll
  for (int t = 0; t < 4; ++t) {
    int col = (cg * 4 + t) * 16 + lm;
#pragma unroll
    for (int r = 0; r < 8; ++r) {
      int row = m0 + rg * 16 + r + 8 * half;
      size_t idx = (size_t)row * UU + col;
      float cc = tanhf(acc[t][r]);
      float u = bf2f(u16[idx]);
      float h = hx[idx];
      ns[idx] = h * u + cc * (1.0f - u);
    }
  }
}

// ------------------------------------------------------- output GEMM ------
// o = [new_state | y_basis] @ w_out + b_out + new_delta   (Ncols=16)
__global__ __launch_bounds__(128) void k_out(
    const float* __restrict__ ns, const float* __restrict__ ybasis,
    const unsigned short* __restrict__ woT, const float* __restrict__ b_out,
    const float* __restrict__ nd, float* __restrict__ o) {
  __shared__ __align__(16) unsigned short sA[64 * KPAD];
  __shared__ __align__(16) unsigned short sB[16 * KPAD];
  const int m0 = blockIdx.x * 64;
  const int tid = threadIdx.x;
  const int lane = tid & 31, wave = tid >> 5;   // wave = row group
  const int lm = lane & 15, half = lane >> 4;

  v8f acc = {};
  for (int k0 = 0; k0 < KK; k0 += 32) {
    // A-tile needs f32->bf16 conversion (ns / y_basis): batch loads first
    float av[16];
#pragma unroll
    for (int j = 0; j < 16; ++j) {
      int e = tid + j * 128, r = e >> 5, k = e & 31, kg = k0 + k;
      float v;
      if (kg < UU)      v = ns[(size_t)(m0 + r) * UU + kg];
      else if (kg < FF) v = ybasis[(size_t)(m0 + r) * DD + (kg - UU)];
      else              v = 0.0f;
      av[j] = v;
    }
    if (tid < 64) {                 // B-tile: 64 chunks from padded woT
      int c = tid >> 2, p = tid & 3;
      cp16_g2l(&sB[c * KPAD + p * 8], &woT[(size_t)c * KK + k0 + p * 8]);
    }
#pragma unroll
    for (int j = 0; j < 16; ++j) {
      int e = tid + j * 128, r = e >> 5, k = e & 31;
      sA[r * KPAD + k] = f2bf(av[j]);
    }
    async_wait0();
    __syncthreads();
    v16bf a = load_frag(&sA[(wave * 16 + lm) * KPAD], half);
    v16bf bf = load_frag(&sB[lm * KPAD], half);
    acc = wmma_bf16(a, bf, acc);
    __syncthreads();
  }
#pragma unroll
  for (int r = 0; r < 8; ++r) {
    int row = m0 + wave * 16 + r + 8 * half;
    size_t idx = (size_t)row * DD + lm;
    o[idx] = acc[r] + b_out[lm] + nd[idx];
  }
}

// ------------------------------------------------------------- launch -----
extern "C" void kernel_launch(void* const* d_in, const int* in_sizes, int n_in,
                              void* d_out, int out_size, void* d_ws, size_t ws_size,
                              hipStream_t stream) {
  (void)in_sizes; (void)n_in; (void)out_size; (void)ws_size;
  const float* x       = (const float*)d_in[0];   // (2,B,N,D)
  const float* hx      = (const float*)d_in[1];   // (B,N*U)
  const float* delta   = (const float*)d_in[2];   // (B,N,D)
  const float* support = (const float*)d_in[3];   // (N,N)
  const float* kr      = (const float*)d_in[4];   // (144,256)
  const float* kc      = (const float*)d_in[5];   // (144,128)
  const float* wout    = (const float*)d_in[6];   // (144,16)
  const float* bout    = (const float*)d_in[7];   // (16)
  const float* wb      = (const float*)d_in[8];   // (16,16)
  const float* bb      = (const float*)d_in[9];   // (16)
  const float* lam     = (const float*)d_in[10];  // (1)

  const float* ybasis = x;
  const float* yres   = x + (size_t)BN * NN * DD;

  char* ws = (char*)d_ws;
  unsigned short* S16  = (unsigned short*)(ws);                 //  2.0 MB
  unsigned short* XS1T = (unsigned short*)(ws + 2097152);       // 37.7 MB
  unsigned short* XS2T = (unsigned short*)(ws + 39845888);      // 37.7 MB
  unsigned short* T16  = (unsigned short*)(ws + 77594624);      // 41.9 MB
  unsigned short* KRT  = (unsigned short*)(ws + 119537664);     // 80 KB
  unsigned short* KCT  = (unsigned short*)(ws + 119619584);     // 40 KB
  unsigned short* WOT  = (unsigned short*)(ws + 119660544);     //  5 KB
  unsigned short* U16  = XS1T;   // alias: xs1T dead after gconv1

  float* o  = (float*)d_out;                       // (B,N,D)
  float* ns = o  + (size_t)MM * DD;                // (B,N*U)
  float* nd = ns + (size_t)MM * UU;                // (B,N,D)

  k_prep_support<<<dim3((NN * NN + 255) / 256), 256, 0, stream>>>(support, S16);
  k_prep_xs<<<dim3((BN * FF * NN + 255) / 256), 256, 0, stream>>>(ybasis, hx, XS1T, XS2T);
  k_prep_weights<<<dim3(((256 + 128 + 16) * KK + 255) / 256), 256, 0, stream>>>(
      kr, kc, wout, KRT, KCT, WOT);
  k_new_delta<<<dim3((MM * DD + 255) / 256), 256, 0, stream>>>(yres, delta, wb, bb, lam, nd);

  k_gconv<<<dim3(NN / 128, BN), 128, 0, stream>>>(S16, XS1T, T16);        // T1
  k_gate_r<<<dim3(MM / 64), 256, 0, stream>>>(T16, KRT, hx, XS2T, U16);
  k_gconv<<<dim3(NN / 128, BN), 128, 0, stream>>>(S16, XS2T, T16);        // T2
  k_gate_c<<<dim3(MM / 64), 256, 0, stream>>>(T16, KCT, hx, U16, ns);
  k_out<<<dim3(MM / 64), 128, 0, stream>>>(ns, ybasis, WOT, bout, nd, o);
}
